// MatchingLayer_72559177499194
// MI455X (gfx1250) — matrix operand order repaired and checked by reading, not compiled, not run on hardware
//
#include <hip/hip_runtime.h>
#include <hip/hip_bf16.h>

#define HB 64
#define SS_ 256
#define HH 256
#define LL 20
#define EPSF 1e-6f

typedef __attribute__((ext_vector_type(16))) _Float16 v16h;
typedef __attribute__((ext_vector_type(8)))  float    v8f;

union Frag { uint4 u[2]; v16h h; };
union H8   { uint4 u; _Float16 h[8]; };

__device__ __forceinline__ v8f wmma_f16(const Frag& a, const Frag& b, v8f c) {
    return __builtin_amdgcn_wmma_f32_16x16x32_f16(
        /*neg_a=*/false, a.h, /*neg_b=*/false, b.h,
        /*c_mod=*/(short)0, c, /*reuse_a=*/false, /*reuse_b=*/false);
}

// ---------------------------------------------------------------------------
// K0: f16 conversion + unweighted inv-norms (cosine matrix) + w_max-weighted
//     inv-norms (max-pooling). One wave per (array, b, s) row.
// array a = d*2 + which (which: 0 = p-half (v1), 1 = q-half (v2))
// ---------------------------------------------------------------------------
__global__ void prep_kernel(const float* __restrict__ p, const float* __restrict__ q,
                            const float* __restrict__ w3, const float* __restrict__ w4,
                            _Float16* __restrict__ v1h, _Float16* __restrict__ v2h,
                            float* __restrict__ invc1, float* __restrict__ invc2,
                            float* __restrict__ rn1, float* __restrict__ rn2)
{
    __shared__ float sq[LL * HH];
    int t = threadIdx.x;
    int rid = blockIdx.x * 8 + (t >> 5);
    int lane = t & 31;
    int a = rid >> 14;             // 0..3, uniform within block
    int d = a >> 1, which = a & 1;
    int r2 = rid & 16383;
    int b = r2 >> 8, s = r2 & 255;

    const float* wm = d ? w4 : w3;
    for (int l = 0; l < LL; l++) { float wv = wm[l * HH + t]; sq[l * HH + t] = wv * wv; }
    __syncthreads();

    const float* src = (which ? q : p) + ((size_t)(b * SS_ + s)) * (2 * HH) + d * HH;
    _Float16* dsth = (which ? v2h : v1h) + ((size_t)d * HB * SS_ + b * SS_ + s) * (size_t)HH;

    int h0 = lane * 8;
    float x2[8]; float ssum = 0.f;
    H8 pk;
    for (int k = 0; k < 8; k++) {
        float x = src[h0 + k];
        x2[k] = x * x; ssum += x2[k];
        pk.h[k] = (_Float16)x;
    }
    *(uint4*)(dsth + h0) = pk.u;

    for (int m = 16; m >= 1; m >>= 1) ssum += __shfl_xor(ssum, m, 32);

    float* rnArr = (which ? rn2 : rn1) + ((size_t)d * HB * SS_ + b * SS_ + s) * LL;
    for (int l = 0; l < LL; l++) {
        float acc = 0.f;
        const float* sl = sq + l * HH + h0;
        for (int k = 0; k < 8; k++) acc += x2[k] * sl[k];
        for (int m = 16; m >= 1; m >>= 1) acc += __shfl_xor(acc, m, 32);
        if (lane == 0) rnArr[l] = 1.0f / sqrtf(fmaxf(acc, EPSF));
    }
    if (lane == 0) {
        float* ic = (which ? invc2 : invc1) + (size_t)d * HB * SS_ + b * SS_ + s;
        *ic = 1.0f / sqrtf(fmaxf(ssum, EPSF));
    }
}

// ---------------------------------------------------------------------------
// Batched 256x256x256 GEMM with f16 WMMA, 64x64 tile per workgroup.
// MODE 0: cm = scale(v1h . v2h^T)  (B source row-major [N][K], direct copy)
// MODE 1: weighted = cmh . v2h     (B source row-major [K][N], transposed stage)
// ---------------------------------------------------------------------------
template<int MODE>
__global__ void gemm_kernel(const _Float16* __restrict__ Aop, const _Float16* __restrict__ Bop,
                            const float* __restrict__ invc1, const float* __restrict__ invc2,
                            float* __restrict__ cm, _Float16* __restrict__ cmh,
                            float* __restrict__ outw)
{
    __shared__ __attribute__((aligned(16))) _Float16 aS[64 * 40];
    __shared__ __attribute__((aligned(16))) _Float16 bS[64 * 40];

    int blk = blockIdx.x;                       // 2*64*4*4 = 2048
    int jT = blk & 3, iT = (blk >> 2) & 3, b = (blk >> 4) & 63, d = blk >> 10;
    size_t batch = (size_t)d * HB + b;
    int i0 = iT * 64, j0 = jT * 64;

    int t = threadIdx.x;
    int wav = t >> 5, lane = t & 31, m = lane & 15, hs = lane >> 4;
    int wj = wav & 3, wi = wav >> 2;            // wi in {0,1}

    const _Float16* Abase = Aop + batch * (size_t)(SS_ * HH);  // 65536 both modes
    const _Float16* Bbase = Bop + batch * (size_t)(SS_ * HH);

    v8f c0 = {}; v8f c1 = {};

    for (int k0 = 0; k0 < 256; k0 += 32) {
        {   // stage A: rows i0..+63, cols k0..+31
            int row = t >> 2, c8 = (t & 3) * 8;
            uint4 va = *(const uint4*)(Abase + (size_t)(i0 + row) * 256 + k0 + c8);
            *(uint4*)(aS + row * 40 + c8) = va;
        }
        if (MODE == 0) {
            int row = t >> 2, c8 = (t & 3) * 8;
            uint4 vb = *(const uint4*)(Bbase + (size_t)(j0 + row) * 256 + k0 + c8);
            *(uint4*)(bS + row * 40 + c8) = vb;
        } else {
            int kk = t >> 3, n8 = (t & 7) * 8;
            H8 vb; vb.u = *(const uint4*)(Bbase + (size_t)(k0 + kk) * 256 + j0 + n8);
            for (int u = 0; u < 8; u++) bS[(n8 + u) * 40 + kk] = vb.h[u];
        }
        __syncthreads();

        Frag fb;
        const uint4* bp = (const uint4*)(bS + (wj * 16 + m) * 40);
        fb.u[0] = bp[hs]; fb.u[1] = bp[2 + hs];
        Frag fa0, fa1;
        const uint4* ap0 = (const uint4*)(aS + (wi * 16 + m) * 40);
        fa0.u[0] = ap0[hs]; fa0.u[1] = ap0[2 + hs];
        const uint4* ap1 = (const uint4*)(aS + (wi * 16 + 32 + m) * 40);
        fa1.u[0] = ap1[hs]; fa1.u[1] = ap1[2 + hs];

        c0 = wmma_f16(fa0, fb, c0);
        c1 = wmma_f16(fa1, fb, c1);
        __syncthreads();
    }

    int j = j0 + wj * 16 + m;                   // N coordinate
    if (MODE == 0) {
        float inj = invc2[batch * SS_ + j];
        float*    cmB  = cm  + batch * (size_t)(SS_ * SS_);
        _Float16* cmhB = cmh + batch * (size_t)(SS_ * SS_);
        for (int r = 0; r < 8; r++) {
            int iA = i0 + wi * 16 + r + 8 * hs;
            float v = c0[r] * invc1[batch * SS_ + iA] * inj;
            cmB[(size_t)iA * SS_ + j] = v;
            cmhB[(size_t)iA * SS_ + j] = (_Float16)v;
            int iB = iA + 32;
            float v2 = c1[r] * invc1[batch * SS_ + iB] * inj;
            cmB[(size_t)iB * SS_ + j] = v2;
            cmhB[(size_t)iB * SS_ + j] = (_Float16)v2;
        }
    } else {
        float* oB = outw + batch * (size_t)(SS_ * HH);
        for (int r = 0; r < 8; r++) {
            int iA = i0 + wi * 16 + r + 8 * hs;
            oB[(size_t)iA * HH + j] = c0[r];
            oB[(size_t)(iA + 32) * HH + j] = c1[r];
        }
    }
}

// ---------------------------------------------------------------------------
// K2b: per-row sum + first-occurrence argmax of cm. One wave per row.
// ---------------------------------------------------------------------------
__global__ void rowstat_kernel(const float* __restrict__ cm,
                               float* __restrict__ rowsum, int* __restrict__ amax)
{
    int rid = blockIdx.x * 8 + (threadIdx.x >> 5);   // 0..32767
    int lane = threadIdx.x & 31;
    const float* row = cm + (size_t)rid * SS_;
    float s = 0.f, best = -3.0e38f; int bi = 0;
    for (int k = 0; k < 8; k++) {
        int j = k * 32 + lane;
        float v = row[j];
        s += v;
        if (v > best) { best = v; bi = j; }
    }
    for (int mm = 16; mm >= 1; mm >>= 1) {
        s += __shfl_xor(s, mm, 32);
        float ob = __shfl_xor(best, mm, 32);
        int   oi = __shfl_xor(bi,   mm, 32);
        if (ob > best || (ob == best && oi < bi)) { best = ob; bi = oi; }
    }
    if (lane == 0) { rowsum[rid] = s; amax[rid] = bi; }
}

// ---------------------------------------------------------------------------
// K3: max-pooling matching. One WG per (d,b,l,i-tile64): A' = v1h*w staged
// once, B'' = v2h*w/n2 streamed, GEMM over K=256, row-max, * 1/n1.
// ---------------------------------------------------------------------------
__global__ void maxpool_kernel(const _Float16* __restrict__ v1h, const _Float16* __restrict__ v2h,
                               const float* __restrict__ w3, const float* __restrict__ w4,
                               const float* __restrict__ rn1, const float* __restrict__ rn2,
                               float* __restrict__ out)
{
    __shared__ __attribute__((aligned(16))) _Float16 aS[64 * 264];
    __shared__ __attribute__((aligned(16))) _Float16 bS[64 * 40];
    __shared__ float wl[HH];
    __shared__ float ml[4 * 64];
    __shared__ float mrun[64];

    int blk = blockIdx.x;                       // ((d*64+b)*20+l)*4 + iT
    int iT = blk & 3;
    int tmp = blk >> 2;
    int l = tmp % 20; tmp /= 20;
    int b = tmp & 63, d = tmp >> 6;
    size_t batch = (size_t)d * HB + b;
    const float* wm = (d ? w4 : w3) + l * HH;

    int t = threadIdx.x;
    int wav = t >> 5, lane = t & 31, m = lane & 15, hs = lane >> 4;
    int wj = wav & 3, wi = wav >> 2;

    wl[t] = wm[t];
    if (t < 64) mrun[t] = -3.0e38f;
    __syncthreads();

    {   // stage full A' = v1h * w  (64 rows x 256)
        int row = t >> 2; int cbase = (t & 3) * 64;
        const _Float16* src = v1h + (batch * SS_ + iT * 64 + row) * (size_t)HH + cbase;
        for (int c = 0; c < 64; c += 8) {
            H8 vv; vv.u = *(const uint4*)(src + c);
            H8 ov;
            for (int u = 0; u < 8; u++) ov.h[u] = (_Float16)((float)vv.h[u] * wl[cbase + c + u]);
            *(uint4*)(aS + row * 264 + cbase + c) = ov.u;
        }
    }
    __syncthreads();

    const float* rn2B = rn2 + batch * (size_t)(SS_ * LL);
    for (int jb = 0; jb < 4; jb++) {
        int j0 = jb * 64;
        v8f c0 = {}; v8f c1 = {};
        for (int k0 = 0; k0 < 256; k0 += 32) {
            __syncthreads();
            {   // stage B'' = v2h * w * (1/n2)
                int row = t >> 2, c8 = (t & 3) * 8;
                float scal = rn2B[(j0 + row) * LL + l];
                H8 vv; vv.u = *(const uint4*)(v2h + (batch * SS_ + j0 + row) * (size_t)HH + k0 + c8);
                H8 ov;
                for (int u = 0; u < 8; u++) ov.h[u] = (_Float16)((float)vv.h[u] * wl[k0 + c8 + u] * scal);
                *(uint4*)(bS + row * 40 + c8) = ov.u;
            }
            __syncthreads();

            Frag fb;
            const uint4* bp = (const uint4*)(bS + (wj * 16 + m) * 40);
            fb.u[0] = bp[hs]; fb.u[1] = bp[2 + hs];
            Frag fa0, fa1;
            const uint4* ap0 = (const uint4*)(aS + (wi * 16 + m) * 264 + k0);
            fa0.u[0] = ap0[hs]; fa0.u[1] = ap0[2 + hs];
            const uint4* ap1 = (const uint4*)(aS + (wi * 16 + 32 + m) * 264 + k0);
            fa1.u[0] = ap1[hs]; fa1.u[1] = ap1[2 + hs];

            c0 = wmma_f16(fa0, fb, c0);
            c1 = wmma_f16(fa1, fb, c1);
        }
        for (int r = 0; r < 8; r++) {
            float v = c0[r];
            for (int mm = 1; mm < 16; mm <<= 1) v = fmaxf(v, __shfl_xor(v, mm, 32));
            if (m == 0) ml[wj * 64 + wi * 16 + r + 8 * hs] = v;
            float v2 = c1[r];
            for (int mm = 1; mm < 16; mm <<= 1) v2 = fmaxf(v2, __shfl_xor(v2, mm, 32));
            if (m == 0) ml[wj * 64 + wi * 16 + 32 + r + 8 * hs] = v2;
        }
        __syncthreads();
        if (t < 64) {
            float mv = fmaxf(fmaxf(ml[t], ml[64 + t]), fmaxf(ml[128 + t], ml[192 + t]));
            mrun[t] = fmaxf(mrun[t], mv);
        }
        __syncthreads();
    }
    if (t < 64) {
        int i = iT * 64 + t;
        float res = mrun[t] * rn1[(batch * SS_ + i) * LL + l];
        out[(size_t)(b * SS_ + i) * 160 + 40 + d * 20 + l] = res;
    }
}

// ---------------------------------------------------------------------------
// K4: full / attentive / max-attentive multi-perspective cosines (fp32 VALU).
// One wave per (d,b,i) row.
// ---------------------------------------------------------------------------
__global__ void persp_kernel(const float* __restrict__ p, const float* __restrict__ q,
                             const float* __restrict__ w1, const float* __restrict__ w2,
                             const float* __restrict__ w5, const float* __restrict__ w6,
                             const float* __restrict__ w7, const float* __restrict__ w8,
                             const float* __restrict__ wgt, const float* __restrict__ rowsum,
                             const int* __restrict__ amax, float* __restrict__ out)
{
    __shared__ float sqF[LL * HH];
    __shared__ float sqA[LL * HH];
    __shared__ float sqM[LL * HH];

    int t = threadIdx.x;
    int rid = blockIdx.x * 8 + (t >> 5);        // [d][b][i]
    int lane = t & 31;
    int d = rid >> 14; int r2 = rid & 16383; int b = r2 >> 8; int i = r2 & 255;

    const float* wF = d ? w2 : w1;
    const float* wA = d ? w6 : w5;
    const float* wM = d ? w8 : w7;
    for (int l = 0; l < LL; l++) {
        float a_ = wF[l * HH + t]; sqF[l * HH + t] = a_ * a_;
        float b_ = wA[l * HH + t]; sqA[l * HH + t] = b_ * b_;
        float c_ = wM[l * HH + t]; sqM[l * HH + t] = c_ * c_;
    }
    __syncthreads();

    size_t rowP = ((size_t)(b * SS_ + i)) * (2 * HH) + d * HH;
    int anchor_s = d ? 0 : (SS_ - 1);
    size_t rowAn = ((size_t)(b * SS_ + anchor_s)) * (2 * HH) + d * HH;
    int idx = amax[rid];
    size_t rowMv = ((size_t)(b * SS_ + idx)) * (2 * HH) + d * HH;
    float inv = 1.0f / (rowsum[rid] + EPSF);
    const float* wrow = wgt + (size_t)rid * HH;

    float p_v11[8], p_van[8], p_ann[8], p_vat[8], p_att[8], p_vmv[8], p_mvv[8];
    for (int k = 0; k < 8; k++) {
        int h = k * 32 + lane;
        float v = p[rowP + h];
        float an = q[rowAn + h];
        float at = wrow[h] * inv;
        float mv = q[rowMv + h];
        p_v11[k] = v * v;  p_van[k] = v * an; p_ann[k] = an * an;
        p_vat[k] = v * at; p_att[k] = at * at;
        p_vmv[k] = v * mv; p_mvv[k] = mv * mv;
    }

    float* orow = out + (size_t)(b * SS_ + i) * 160;
    for (int l = 0; l < LL; l++) {
        float aF = 0, bF = 0, cF = 0, aA = 0, bA = 0, cA = 0, aM = 0, bM = 0, cM = 0;
        for (int k = 0; k < 8; k++) {
            int h = k * 32 + lane;
            float sf = sqF[l * HH + h], sa = sqA[l * HH + h], sm = sqM[l * HH + h];
            aF += p_van[k] * sf; bF += p_v11[k] * sf; cF += p_ann[k] * sf;
            aA += p_vat[k] * sa; bA += p_v11[k] * sa; cA += p_att[k] * sa;
            aM += p_vmv[k] * sm; bM += p_v11[k] * sm; cM += p_mvv[k] * sm;
        }
        for (int mm = 16; mm >= 1; mm >>= 1) {
            aF += __shfl_xor(aF, mm, 32); bF += __shfl_xor(bF, mm, 32); cF += __shfl_xor(cF, mm, 32);
            aA += __shfl_xor(aA, mm, 32); bA += __shfl_xor(bA, mm, 32); cA += __shfl_xor(cA, mm, 32);
            aM += __shfl_xor(aM, mm, 32); bM += __shfl_xor(bM, mm, 32); cM += __shfl_xor(cM, mm, 32);
        }
        if (lane == 0) {
            orow[0   + d * 20 + l] = aF / (sqrtf(fmaxf(bF, EPSF)) * sqrtf(fmaxf(cF, EPSF)));
            orow[80  + d * 20 + l] = aA / (sqrtf(fmaxf(bA, EPSF)) * sqrtf(fmaxf(cA, EPSF)));
            orow[120 + d * 20 + l] = aM / (sqrtf(fmaxf(bM, EPSF)) * sqrtf(fmaxf(cM, EPSF)));
        }
    }
}

// ---------------------------------------------------------------------------
extern "C" void kernel_launch(void* const* d_in, const int* in_sizes, int n_in,
                              void* d_out, int out_size, void* d_ws, size_t ws_size,
                              hipStream_t stream) {
    const float* p  = (const float*)d_in[0];
    const float* q  = (const float*)d_in[1];
    const float* w1 = (const float*)d_in[2];
    const float* w2 = (const float*)d_in[3];
    const float* w3 = (const float*)d_in[4];
    const float* w4 = (const float*)d_in[5];
    const float* w5 = (const float*)d_in[6];
    const float* w6 = (const float*)d_in[7];
    const float* w7 = (const float*)d_in[8];
    const float* w8 = (const float*)d_in[9];
    float* out = (float*)d_out;

    const size_t BS  = (size_t)HB * SS_;        // 16384
    const size_t BSH = BS * HH;                 // 4.19M
    const size_t BSS = BS * SS_;                // 4.19M
    const size_t BSL = BS * LL;

    char* w = (char*)d_ws;
    auto alloc = [&](size_t bytes) -> void* {
        void* r = (void*)w; w += (bytes + 255) & ~(size_t)255; return r;
    };
    _Float16* v1h    = (_Float16*)alloc(2 * BSH * sizeof(_Float16));
    _Float16* v2h    = (_Float16*)alloc(2 * BSH * sizeof(_Float16));
    float*    invc1  = (float*)alloc(2 * BS * sizeof(float));
    float*    invc2  = (float*)alloc(2 * BS * sizeof(float));
    float*    rn1    = (float*)alloc(2 * BSL * sizeof(float));
    float*    rn2    = (float*)alloc(2 * BSL * sizeof(float));
    float*    cm     = (float*)alloc(2 * BSS * sizeof(float));
    _Float16* cmh    = (_Float16*)alloc(2 * BSS * sizeof(_Float16));
    float*    rowsum = (float*)alloc(2 * BS * sizeof(float));
    int*      amax   = (int*)alloc(2 * BS * sizeof(int));
    float*    wgt    = (float*)alloc(2 * BSH * sizeof(float));
    (void)in_sizes; (void)n_in; (void)out_size; (void)ws_size;

    prep_kernel<<<8192, 256, 0, stream>>>(p, q, w3, w4, v1h, v2h, invc1, invc2, rn1, rn2);
    gemm_kernel<0><<<2048, 256, 0, stream>>>(v1h, v2h, invc1, invc2, cm, cmh, nullptr);
    rowstat_kernel<<<4096, 256, 0, stream>>>(cm, rowsum, amax);
    gemm_kernel<1><<<2048, 256, 0, stream>>>(cmh, v2h, nullptr, nullptr, nullptr, nullptr, wgt);
    maxpool_kernel<<<10240, 256, 0, stream>>>(v1h, v2h, w3, w4, rn1, rn2, out);
    persp_kernel<<<4096, 256, 0, stream>>>(p, q, w1, w2, w5, w6, w7, w8, wgt, rowsum, amax, out);
}
